// SAE_61495341744340
// MI455X (gfx1250) — matrix operand order repaired and checked by reading, not compiled, not run on hardware
//
#include <hip/hip_runtime.h>

#define N_TOK   16384
#define D_MODEL 2048
#define D_FEAT  8192
#define K_TOP   30

typedef __attribute__((ext_vector_type(16))) __bf16 v16bf;
typedef __attribute__((ext_vector_type(8)))  float  v8f;

union FragU { v16bf bf; uint4 q[2]; };

__device__ __forceinline__ unsigned short f32_to_bf16_rne(float f) {
    unsigned u = __float_as_uint(f);
    unsigned r = u + 0x7FFFu + ((u >> 16) & 1u);
    return (unsigned short)(r >> 16);
}
__device__ __forceinline__ float bf16_bits_to_f32(unsigned short h) {
    return __uint_as_float(((unsigned)h) << 16);
}

// ---------------- conversion: x - b_dec -> (hi, lo) bf16 split ----------------
__global__ void sae_cvt_x_split(const float* __restrict__ x, const float* __restrict__ b_dec,
                                unsigned short* __restrict__ a_hi, unsigned short* __restrict__ a_lo) {
    size_t i = (size_t)blockIdx.x * blockDim.x + threadIdx.x;
    if (i >= (size_t)N_TOK * D_MODEL) return;
    int d = (int)(i & (D_MODEL - 1));
    float v = x[i] - b_dec[d];
    unsigned short hi = f32_to_bf16_rne(v);
    float vhi = bf16_bits_to_f32(hi);
    a_hi[i] = hi;
    a_lo[i] = f32_to_bf16_rne(v - vhi);
}

__global__ void sae_cvt_w_split(const float* __restrict__ w,
                                unsigned short* __restrict__ b_hi, unsigned short* __restrict__ b_lo) {
    size_t i = (size_t)blockIdx.x * blockDim.x + threadIdx.x;
    if (i >= (size_t)D_FEAT * D_MODEL) return;
    float v = w[i];
    unsigned short hi = f32_to_bf16_rne(v);
    float vhi = bf16_bits_to_f32(hi);
    b_hi[i] = hi;
    b_lo[i] = f32_to_bf16_rne(v - vhi);
}

// ---------------- w_dec [D_MODEL, D_FEAT] -> w_decT [D_FEAT, D_MODEL] ----------------
__global__ void sae_transpose_wdec(const float* __restrict__ wdec, float* __restrict__ wdecT) {
    __shared__ float tile[32][33];
    int f0 = blockIdx.x * 32;
    int d0 = blockIdx.y * 32;
    int tx = threadIdx.x, ty = threadIdx.y;        // block (32, 8)
#pragma unroll
    for (int r = 0; r < 32; r += 8)
        tile[ty + r][tx] = wdec[(size_t)(d0 + ty + r) * D_FEAT + (f0 + tx)];
    __syncthreads();
#pragma unroll
    for (int r = 0; r < 32; r += 8)
        wdecT[(size_t)(f0 + ty + r) * D_MODEL + (d0 + tx)] = tile[tx][ty + r];
}

// ---------------- encode GEMM: x_hid = Asplit * Bsplit^T (bf16x3, f32 acc) ----------------
// LDS staging: 4 tiles (A-hi, A-lo, B-hi, B-lo), each 128 rows x 64B data, row stride 80B
#define TILE_STRIDE 80
#define TILE_BYTES  (128 * TILE_STRIDE)   // 10240
#define BUF_BYTES   (4 * TILE_BYTES)      // 40960 per stage buffer

// async copy of one 128x32-bf16 tile (128 x 64B) from global into padded LDS tile
__device__ __forceinline__ void async_tile_128x32(const unsigned short* gbase, int row0, int k0,
                                                  char* ltile, int tid) {
#pragma unroll
    for (int h = 0; h < 2; ++h) {
        int o   = (tid << 4) + (h << 12);      // 16B chunk offset within 8KB of data
        int row = o >> 6;
        int col = o & 63;
        unsigned long long g = (unsigned long long)(size_t)
            ((const char*)(gbase + (size_t)(row0 + row) * D_MODEL + k0) + col);
        unsigned l = (unsigned)(size_t)(ltile + row * TILE_STRIDE + col);  // low 32b = LDS offset
        asm volatile("global_load_async_to_lds_b128 %0, %1, off"
                     :: "v"(l), "v"(g) : "memory");
    }
}

// A fragment (16x32 bf16, MxK): lane<16 holds K{0..7,16..23}; lane>=16 holds K{8..15,24..31}
__device__ __forceinline__ v16bf lds_fragA(const char* tile, int row0, int lane) {
    const char* p = tile + (row0 + (lane & 15)) * TILE_STRIDE + ((lane >> 4) << 4);
    FragU u;
    u.q[0] = *(const uint4*)p;
    u.q[1] = *(const uint4*)(p + 32);
    return u.bf;
}
// B fragment (32x16 bf16, KxN): lane n holds col n&15; lane<16 holds K{0..7,16..23}; lane>=16 holds K{8..15,24..31}
__device__ __forceinline__ v16bf lds_fragB(const char* tile, int row0, int lane) {
    const char* p = tile + (row0 + (lane & 15)) * TILE_STRIDE + ((lane >> 4) << 4);
    FragU u;
    u.q[0] = *(const uint4*)p;
    u.q[1] = *(const uint4*)(p + 32);
    return u.bf;
}

__global__ void __launch_bounds__(256)
sae_encode_gemm(const unsigned short* __restrict__ a_hi, const unsigned short* __restrict__ a_lo,
                const unsigned short* __restrict__ b_hi, const unsigned short* __restrict__ b_lo,
                float* __restrict__ xhid) {
    extern __shared__ char smem[];                 // 2 * BUF_BYTES
    const int tid  = threadIdx.x;
    const int lane = tid & 31;
    const int wave = tid >> 5;
    const int wm = wave >> 2, wn = wave & 3;       // wave tile 64(M) x 32(N)
    const int mblk = blockIdx.y * 128;
    const int fblk = blockIdx.x * 128;

    v8f acc[4][2];
#pragma unroll
    for (int i = 0; i < 4; ++i)
#pragma unroll
        for (int j = 0; j < 2; ++j) acc[i][j] = (v8f){};

    // prologue: stage k-tile 0 into buffer 0
    async_tile_128x32(a_hi, mblk, 0, smem, tid);
    async_tile_128x32(a_lo, mblk, 0, smem + TILE_BYTES, tid);
    async_tile_128x32(b_hi, fblk, 0, smem + 2 * TILE_BYTES, tid);
    async_tile_128x32(b_lo, fblk, 0, smem + 3 * TILE_BYTES, tid);

    const int NSTEP = D_MODEL / 32;
    for (int t = 0; t < NSTEP; ++t) {
        asm volatile("s_wait_asynccnt 0x0" ::: "memory");
        __syncthreads();                           // buffer t published to all waves
        char* cur = smem + (t & 1) * BUF_BYTES;
        if (t + 1 < NSTEP) {                       // overlap copies of t+1 with compute on t
            char* nxt = smem + ((t + 1) & 1) * BUF_BYTES;
            const int k0 = (t + 1) * 32;
            async_tile_128x32(a_hi, mblk, k0, nxt, tid);
            async_tile_128x32(a_lo, mblk, k0, nxt + TILE_BYTES, tid);
            async_tile_128x32(b_hi, fblk, k0, nxt + 2 * TILE_BYTES, tid);
            async_tile_128x32(b_lo, fblk, k0, nxt + 3 * TILE_BYTES, tid);
        }
        const char* Ah = cur;
        const char* Al = cur + TILE_BYTES;
        const char* Bh = cur + 2 * TILE_BYTES;
        const char* Bl = cur + 3 * TILE_BYTES;

        v16bf ah[4], al[4], bh[2], bl[2];
#pragma unroll
        for (int i = 0; i < 4; ++i) {
            ah[i] = lds_fragA(Ah, wm * 64 + i * 16, lane);
            al[i] = lds_fragA(Al, wm * 64 + i * 16, lane);
        }
#pragma unroll
        for (int j = 0; j < 2; ++j) {
            bh[j] = lds_fragB(Bh, wn * 32 + j * 16, lane);
            bl[j] = lds_fragB(Bl, wn * 32 + j * 16, lane);
        }
#pragma unroll
        for (int i = 0; i < 4; ++i)
#pragma unroll
            for (int j = 0; j < 2; ++j) {
                acc[i][j] = __builtin_amdgcn_wmma_f32_16x16x32_bf16(
                    false, ah[i], false, bh[j], (short)0, acc[i][j], false, false);
                acc[i][j] = __builtin_amdgcn_wmma_f32_16x16x32_bf16(
                    false, ah[i], false, bl[j], (short)0, acc[i][j], false, false);
                acc[i][j] = __builtin_amdgcn_wmma_f32_16x16x32_bf16(
                    false, al[i], false, bh[j], (short)0, acc[i][j], false, false);
            }
    }

    const int rlo = (lane >> 4) << 3;
    const int col = lane & 15;
#pragma unroll
    for (int i = 0; i < 4; ++i)
#pragma unroll
        for (int j = 0; j < 2; ++j)
#pragma unroll
            for (int r = 0; r < 8; ++r)
                xhid[(size_t)(mblk + wm * 64 + i * 16 + rlo + r) * D_FEAT
                     + (fblk + wn * 32 + j * 16 + col)] = acc[i][j][r];
}

// ---------------- top-K radix select + in-place mask ----------------
__global__ void __launch_bounds__(256)
sae_topk_mask(float* __restrict__ xhid, float* __restrict__ out_val, int* __restrict__ out_idx) {
    __shared__ unsigned keys[D_FEAT];
    __shared__ unsigned hist[256];
    __shared__ unsigned s_chosen, s_need;
    __shared__ int cnt_eq, out_cnt;
    const int row = blockIdx.x;
    const int tid = threadIdx.x;
    float* rowp = xhid + (size_t)row * D_FEAT;

    for (int i = tid; i < D_FEAT; i += 256) {
        unsigned u = __float_as_uint(rowp[i]);
        keys[i] = (u & 0x80000000u) ? ~u : (u | 0x80000000u);   // order-preserving map
    }
    if (tid == 0) { cnt_eq = 0; out_cnt = 0; }
    __syncthreads();

    unsigned prefix = 0, pmask = 0, need = K_TOP;
#pragma unroll
    for (int level = 0; level < 4; ++level) {
        const int shift = 24 - 8 * level;
        hist[tid] = 0;
        __syncthreads();
        for (int i = tid; i < D_FEAT; i += 256) {
            unsigned k = keys[i];
            if ((k & pmask) == prefix) atomicAdd(&hist[(k >> shift) & 255u], 1u);
        }
        __syncthreads();
        if (tid == 0) {
            unsigned cum = 0; int b = 255;
            for (; b >= 0; --b) {
                unsigned h = hist[b];
                if (cum + h >= need) break;
                cum += h;
            }
            if (b < 0) b = 0;
            s_chosen = (unsigned)b;
            s_need = need - cum;
        }
        __syncthreads();
        prefix |= (s_chosen << shift);
        pmask  |= (0xFFu << shift);
        need = s_need;
        __syncthreads();
    }

    const unsigned T = prefix;   // key of the K-th largest value
    for (int i = tid; i < D_FEAT; i += 256) {
        unsigned k = keys[i];
        bool sel = false;
        if (k > T) sel = true;
        else if (k == T) { int c = atomicAdd(&cnt_eq, 1); if ((unsigned)c < need) sel = true; }
        float v = (k & 0x80000000u) ? __uint_as_float(k & 0x7FFFFFFFu) : __uint_as_float(~k);
        rowp[i] = sel ? v : 0.0f;
        if (sel) {
            int o = atomicAdd(&out_cnt, 1);
            out_val[row * 32 + o] = v;
            out_idx[row * 32 + o] = i;
        }
    }
}

// ---------------- sparse decode: x_hat = sum_j v_j * w_decT[idx_j,:] + b_dec ----------------
__global__ void __launch_bounds__(256)
sae_decode(const float* __restrict__ out_val, const int* __restrict__ out_idx,
           const float* __restrict__ wdecT, const float* __restrict__ b_dec,
           float* __restrict__ xhat) {
    __shared__ float sv[K_TOP];
    __shared__ int   si[K_TOP];
    const int row = blockIdx.x;
    const int tid = threadIdx.x;
    if (tid < K_TOP) { sv[tid] = out_val[row * 32 + tid]; si[tid] = out_idx[row * 32 + tid]; }
    __syncthreads();
#pragma unroll
    for (int rep = 0; rep < D_MODEL / 256; ++rep) {
        int d = tid + rep * 256;
        float acc = b_dec[d];
#pragma unroll 5
        for (int j = 0; j < K_TOP; ++j)
            acc += sv[j] * wdecT[(size_t)si[j] * D_MODEL + d];
        xhat[(size_t)row * D_MODEL + d] = acc;
    }
}

extern "C" void kernel_launch(void* const* d_in, const int* in_sizes, int n_in,
                              void* d_out, int out_size, void* d_ws, size_t ws_size,
                              hipStream_t stream) {
    const float* x     = (const float*)d_in[0];
    const float* w_enc = (const float*)d_in[1];
    const float* w_dec = (const float*)d_in[2];
    const float* b_dec = (const float*)d_in[3];

    float* xhat = (float*)d_out;                              // [N_TOK, D_MODEL]
    float* xhid = xhat + (size_t)N_TOK * D_MODEL;             // [N_TOK, D_FEAT]

    char* ws = (char*)d_ws;
    size_t off = 0;
    unsigned short* a_hi = (unsigned short*)(ws + off); off += (size_t)N_TOK * D_MODEL * 2;
    unsigned short* a_lo = (unsigned short*)(ws + off); off += (size_t)N_TOK * D_MODEL * 2;
    unsigned short* b_hi = (unsigned short*)(ws + off); off += (size_t)D_FEAT * D_MODEL * 2;
    unsigned short* b_lo = (unsigned short*)(ws + off); off += (size_t)D_FEAT * D_MODEL * 2;
    float* wdecT  = (float*)(ws + off); off += (size_t)D_FEAT * D_MODEL * 4;
    float* tk_val = (float*)(ws + off); off += (size_t)N_TOK * 32 * 4;
    int*   tk_idx = (int*)(ws + off);   off += (size_t)N_TOK * 32 * 4;

    {   // split-convert inputs
        size_t tot = (size_t)N_TOK * D_MODEL;
        sae_cvt_x_split<<<dim3((unsigned)((tot + 255) / 256)), dim3(256), 0, stream>>>(x, b_dec, a_hi, a_lo);
        size_t totw = (size_t)D_FEAT * D_MODEL;
        sae_cvt_w_split<<<dim3((unsigned)((totw + 255) / 256)), dim3(256), 0, stream>>>(w_enc, b_hi, b_lo);
    }
    sae_transpose_wdec<<<dim3(D_FEAT / 32, D_MODEL / 32), dim3(32, 8), 0, stream>>>(w_dec, wdecT);
    sae_encode_gemm<<<dim3(D_FEAT / 128, N_TOK / 128), dim3(256), 2 * BUF_BYTES, stream>>>(
        a_hi, a_lo, b_hi, b_lo, xhid);
    sae_topk_mask<<<dim3(N_TOK), dim3(256), 0, stream>>>(xhid, tk_val, tk_idx);
    sae_decode<<<dim3(N_TOK), dim3(256), 0, stream>>>(tk_val, tk_idx, wdecT, b_dec, xhat);
}
